// RTRBM_27023934227031
// MI455X (gfx1250) — matrix operand order repaired and checked by reading, not compiled, not run on hardware
//
#include <hip/hip_runtime.h>

typedef __attribute__((ext_vector_type(16))) _Float16 v16h;
typedef __attribute__((ext_vector_type(8)))  _Float16 v8h;
typedef __attribute__((ext_vector_type(4)))  _Float16 v4h;
typedef __attribute__((ext_vector_type(8)))  float    v8f;
typedef __attribute__((ext_vector_type(4)))  unsigned v4u;
typedef __attribute__((ext_vector_type(8)))  unsigned v8u;

#define NV_ 1024
#define NH_ 512
#define TT_ 128
#define BB_ 256
#define TB_ (TT_ * BB_)   // 32768

__device__ __forceinline__ float sigmoidf_(float x) {
  return 1.0f / (1.0f + __expf(-x));
}

// Deterministic counter-based uniform in [0,1)
__device__ __forceinline__ float urand01(unsigned x, unsigned seed) {
  x ^= seed * 0x9E3779B9u;
  x *= 0x85EBCA6Bu; x ^= x >> 13;
  x *= 0xC2B2AE35u; x ^= x >> 16;
  return (float)(x >> 8) * (1.0f / 16777216.0f);
}

__device__ __forceinline__ v16h cat8(v8h lo, v8h hi) {
  return __builtin_shufflevector(lo, hi, 0,1,2,3,4,5,6,7,8,9,10,11,12,13,14,15);
}

// ---------------------------------------------------------------------------
// Tiled WMMA GEMM: C(MxN, f32 acc) = A(MxK, f16 row-major) * B(KxN, f16 row-major)
// Block tile 128x128, K-chunk 32. 256 threads = 8 waves (4 along M, 2 along N),
// each wave computes a 32x64 tile via 2x4 v_wmma_f32_16x16x32_f16 accumulators.
// A tile is moved global->LDS by the Tensor Data Mover (tensor_load_to_lds,
// TENSORcnt); B tile is staged transposed by the waves (VMEM->LDS).
//   MODE 0: C32 = raw f32 result
//   MODE 1: s = bernoulli(sigmoid(acc + bias0[m])); store s to C16 and/or C32
//   MODE 3: val = sigmoid(acc + add[m,n] + bias0[m]); store f32 to C32, f16 to C16
// ---------------------------------------------------------------------------
template <int MODE>
__global__ __launch_bounds__(256)
void gemm_wmma(const _Float16* __restrict__ A, int lda,
               const _Float16* __restrict__ Bm, int ldb,
               float* __restrict__ C32, int ldc32,
               _Float16* __restrict__ C16, int ldc16,
               const float* __restrict__ add, int ldadd,
               const float* __restrict__ bias0,
               int M, int N, int K, unsigned seed)
{
  __shared__ alignas(64) _Float16 As [128 * 32];  // [m][k]  (filled by TDM)
  __shared__ alignas(64) _Float16 Bst[128 * 32];  // [n][k]  (B transposed in LDS)

  const int tid     = threadIdx.x;
  const int lane    = tid & 31;
  const int wave    = tid >> 5;
  const int wm      = wave & 3;          // 4 waves along M  (32 rows each)
  const int wn      = wave >> 2;         // 2 waves along N  (64 cols each)
  const int m0      = blockIdx.y * 128;
  const int n0      = blockIdx.x * 128;
  const int half_id = lane >> 4;
  const int l16     = lane & 15;

  // LDS byte address of As for the TDM descriptor (low 32 bits of generic ptr)
  const unsigned ldsAs = (unsigned)(unsigned long long)(const void*)&As[0];

  v8f acc[2][4];
  const v8f vzero = {0.f,0.f,0.f,0.f,0.f,0.f,0.f,0.f};
#pragma unroll
  for (int mi = 0; mi < 2; ++mi)
#pragma unroll
    for (int ni = 0; ni < 4; ++ni) acc[mi][ni] = vzero;

  for (int kc = 0; kc < K; kc += 32) {
    // ---- A tile 128x32 via Tensor Data Mover (async DMA global->LDS) ----
    // D# per CDNA5 ISA ch.8: 2D tile, tile_dim0=32 (contiguous K),
    // tile_dim1=128 rows, row stride = lda elements, data_size=2B.
    if (wave == 0) {
      unsigned long long ga =
          (unsigned long long)(const void*)(A + (size_t)m0 * lda + kc);
      const unsigned td0 = 0x40000000u, td1 = 0x40000000u;  // huge: no OOB clip
      v4u g0 = { 1u,                                    // count=1, user mode
                 ldsAs,                                 // lds_addr
                 (unsigned)ga,                          // global_addr[31:0]
                 (unsigned)((ga >> 32) & 0x01FFFFFFu)   // global_addr[56:32]
                     | (2u << 30) };                    // type=2 (image)
      v8u g1 = { 0x00010000u,                           // wg_mask=0, data_size=2B
                 (td0 & 0xFFFFu) << 16,                 // tensor_dim0[15:0]
                 (td0 >> 16) | ((td1 & 0xFFFFu) << 16), // td0[31:16] | td1[15:0]
                 (td1 >> 16) | (32u << 16),             // td1[31:16] | tile_dim0=32
                 128u,                                  // tile_dim1=128, tile_dim2=0
                 (unsigned)lda,                         // tensor_dim0_stride[31:0]
                 0u, 0u };                              // stride hi, dim1_stride=0
      v4u gz = { 0u, 0u, 0u, 0u };                      // groups 2/3: 2D tensor
      asm volatile("tensor_load_to_lds %0, %1, %2, %3"
                   :: "s"(g0), "s"(g1), "s"(gz), "s"(gz) : "memory");
    }

    // ---- Stage B tile 32x128 -> LDS transposed (coalesced 8B loads) ----
#pragma unroll
    for (int g = tid; g < 1024; g += 256) {
      int k  = g >> 5;
      int n4 = (g & 31) << 2;
      v4h d = *(const v4h*)(Bm + (size_t)(kc + k) * ldb + (n0 + n4));
#pragma unroll
      for (int i = 0; i < 4; ++i) Bst[(n4 + i) * 32 + k] = d[i];
    }
    if (kc + 32 < K) {  // global_prefetch_b8 for next B K-chunk
      __builtin_prefetch(Bm + (size_t)(kc + 32 + (tid >> 3)) * ldb + n0, 0, 0);
    }
    if (wave == 0) __builtin_amdgcn_s_wait_tensorcnt(0);  // TDM tile landed
    __syncthreads();

    // A fragment (16x32 f16): lanes 0-15 hold K=[0..7],[16..23]; lanes 16-31 K=[8..15],[24..31]
    v16h afrag[2];
#pragma unroll
    for (int mi = 0; mi < 2; ++mi) {
      int mrow = wm * 32 + mi * 16 + l16;
      int kb   = half_id * 8;
      afrag[mi] = cat8(*(const v8h*)&As[mrow * 32 + kb],
                       *(const v8h*)&As[mrow * 32 + 16 + kb]);
    }
    // B fragment (32x16 f16): lane%16 = N col; lanes 0-15 K=[0..7],[16..23]; lanes 16-31 K=[8..15],[24..31]
    v16h bfrag[4];
#pragma unroll
    for (int ni = 0; ni < 4; ++ni) {
      int ncol = wn * 64 + ni * 16 + l16;
      bfrag[ni] = cat8(*(const v8h*)&Bst[ncol * 32 + half_id * 8],
                       *(const v8h*)&Bst[ncol * 32 + 16 + half_id * 8]);
    }
#pragma unroll
    for (int mi = 0; mi < 2; ++mi)
#pragma unroll
      for (int ni = 0; ni < 4; ++ni)
        acc[mi][ni] = __builtin_amdgcn_wmma_f32_16x16x32_f16(
            false, afrag[mi], false, bfrag[ni], (short)0, acc[mi][ni],
            false, false);
    __syncthreads();
  }

  // Epilogue: C/D layout: VGPR r -> M = r + 8*half_id, N = lane%16
#pragma unroll
  for (int mi = 0; mi < 2; ++mi) {
#pragma unroll
    for (int ni = 0; ni < 4; ++ni) {
#pragma unroll
      for (int r = 0; r < 8; ++r) {
        int gm = m0 + wm * 32 + mi * 16 + r + 8 * half_id;
        int gn = n0 + wn * 64 + ni * 16 + l16;
        float val = acc[mi][ni][r];
        if (MODE == 0) {
          C32[(size_t)gm * ldc32 + gn] = val;
        } else if (MODE == 1) {
          float p = sigmoidf_(val + bias0[gm]);
          float s = (urand01((unsigned)((size_t)gm * N + gn), seed) < p) ? 1.0f : 0.0f;
          if (C16) C16[(size_t)gm * ldc16 + gn] = (_Float16)s;
          if (C32) C32[(size_t)gm * ldc32 + gn] = s;
        } else {  // MODE 3: recurrence step
          float vv = sigmoidf_(val + add[(size_t)gm * ldadd + gn] + bias0[gm]);
          C32[(size_t)gm * ldc32 + gn] = vv;
          C16[(size_t)gm * ldc16 + gn] = (_Float16)vv;
        }
      }
    }
  }
}

// ---------------------------------------------------------------------------
// Elementwise helpers
// ---------------------------------------------------------------------------
__global__ void convert_W_kernel(const float* __restrict__ W,
                                 _Float16* __restrict__ W16,
                                 _Float16* __restrict__ WT16) {
  int i = blockIdx.x * 256 + threadIdx.x;   // over H*V
  int h = i >> 10;                          // / N_V
  int v = i & (NV_ - 1);
  float w = W[i];
  W16[i] = (_Float16)w;
  WT16[(size_t)v * NH_ + h] = (_Float16)w;
}

__global__ void convert_U_kernel(const float* __restrict__ U, _Float16* __restrict__ U16) {
  int i = blockIdx.x * 256 + threadIdx.x;
  U16[i] = (_Float16)U[i];
}

__global__ void convert_v_kernel(const float* __restrict__ v, _Float16* __restrict__ v16) {
  size_t i = (size_t)blockIdx.x * 256 + threadIdx.x;
  v16[i] = (_Float16)v[i];
}

__global__ void r0_kernel(const float* __restrict__ Wv, const float* __restrict__ b_init,
                          float* __restrict__ r32, _Float16* __restrict__ r16) {
  int i = blockIdx.x * 256 + threadIdx.x;   // over H*B
  int h = i >> 8;
  int b = i & (BB_ - 1);
  size_t idx = (size_t)h * TB_ + b;
  float val = sigmoidf_(Wv[idx] + b_init[h]);
  r32[idx] = val;
  r16[idx] = (_Float16)val;
}

// h = bernoulli(parallel_r): col c<B uses b_init, else Ur shifted by B cols + b_h
__global__ void hsample_kernel(const float* __restrict__ Wv, const float* __restrict__ Ur,
                               const float* __restrict__ b_init, const float* __restrict__ b_h,
                               _Float16* __restrict__ h16, unsigned seed) {
  size_t i = (size_t)blockIdx.x * 256 + threadIdx.x;  // over H*TB
  int h = (int)(i >> 15);                             // / TB
  int c = (int)(i & (TB_ - 1));
  float acc = Wv[i];
  float p = (c < BB_) ? sigmoidf_(acc + b_init[h])
                      : sigmoidf_(acc + Ur[i - BB_] + b_h[h]);
  float s = (urand01((unsigned)i, seed) < p) ? 1.0f : 0.0f;
  h16[i] = (_Float16)s;
}

// ---------------------------------------------------------------------------
extern "C" void kernel_launch(void* const* d_in, const int* in_sizes, int n_in,
                              void* d_out, int out_size, void* d_ws, size_t ws_size,
                              hipStream_t stream) {
  (void)in_sizes; (void)n_in; (void)out_size; (void)ws_size;
  const float* v_data = (const float*)d_in[0];   // (V,T,B)
  const float* W      = (const float*)d_in[1];   // (H,V)
  const float* U      = (const float*)d_in[2];   // (H,H)
  const float* b_h    = (const float*)d_in[3];
  const float* b_init = (const float*)d_in[4];
  const float* b_v    = (const float*)d_in[5];
  const int CDk = 2;   // reference uses concrete Python int CDk=2

  float* v_model_out = (float*)d_out;                        // (V,T,B) f32
  float* r_out       = (float*)d_out + (size_t)NV_ * TB_;    // (H,T,B) f32

  // Workspace carve
  char* p = (char*)d_ws;
  auto carve = [&](size_t bytes) { char* q = p; p += (bytes + 255) & ~(size_t)255; return q; };
  _Float16* W16  = (_Float16*)carve((size_t)NH_ * NV_ * 2);
  _Float16* WT16 = (_Float16*)carve((size_t)NV_ * NH_ * 2);
  _Float16* U16  = (_Float16*)carve((size_t)NH_ * NH_ * 2);
  _Float16* v16  = (_Float16*)carve((size_t)NV_ * TB_ * 2);
  float*    Wv   = (float*)   carve((size_t)NH_ * TB_ * 4);  // reused for W@v_model
  float*    Ur   = (float*)   carve((size_t)NH_ * TB_ * 4);
  _Float16* r16  = (_Float16*)carve((size_t)NH_ * TB_ * 2);
  _Float16* h16  = (_Float16*)carve((size_t)NH_ * TB_ * 2);
  _Float16* vm16 = (_Float16*)carve((size_t)NV_ * TB_ * 2);

  // Pack f32 -> f16 operands (v is exactly 0/1: lossless)
  convert_W_kernel<<<(NH_ * NV_) / 256, 256, 0, stream>>>(W, W16, WT16);
  convert_U_kernel<<<(NH_ * NH_) / 256, 256, 0, stream>>>(U, U16);
  convert_v_kernel<<<(int)(((size_t)NV_ * TB_) / 256), 256, 0, stream>>>(v_data, v16);

  // Wv = W @ v  (512 x 32768 x 1024)
  gemm_wmma<0><<<dim3(TB_ / 128, NH_ / 128), 256, 0, stream>>>(
      W16, NV_, v16, TB_, Wv, TB_, nullptr, 0, nullptr, 0, nullptr,
      NH_, TB_, NV_, 0u);

  // r_0 = sigmoid(Wv_0 + b_init)
  r0_kernel<<<(NH_ * BB_) / 256, 256, 0, stream>>>(Wv, b_init, r_out, r16);

  // Serial recurrence: r_t = sigmoid(Wv_t + U @ r_{t-1} + b_h)
  for (int t = 1; t < TT_; ++t) {
    gemm_wmma<3><<<dim3(BB_ / 128, NH_ / 128), 256, 0, stream>>>(
        U16, NH_, r16 + (size_t)(t - 1) * BB_, TB_,
        r_out + (size_t)t * BB_, TB_, r16 + (size_t)t * BB_, TB_,
        Wv + (size_t)t * BB_, TB_, b_h, NH_, BB_, NH_, 0u);
  }

  // Ur = U @ r (512 x 32768 x 512); consumed shifted by B columns everywhere
  gemm_wmma<0><<<dim3(TB_ / 128, NH_ / 128), 256, 0, stream>>>(
      U16, NH_, r16, TB_, Ur, TB_, nullptr, 0, nullptr, 0, nullptr,
      NH_, TB_, NH_, 0u);

  // h_model #1 (elementwise: Wv already holds W@v_data)
  hsample_kernel<<<(int)(((size_t)NH_ * TB_) / 256), 256, 0, stream>>>(
      Wv, Ur, b_init, b_h, h16, 0x12345u);

  // CD-k Gibbs chain (CDk-1 iterations)
  unsigned seed = 0x9000u;
  for (int it = 0; it < CDk - 1; ++it) {
    // v_model = bernoulli(sigmoid(W^T @ h + b_v))  (fused epilogue, f16 out)
    gemm_wmma<1><<<dim3(TB_ / 128, NV_ / 128), 256, 0, stream>>>(
        WT16, NH_, h16, TB_, nullptr, 0, vm16, TB_, nullptr, 0, b_v,
        NV_, TB_, NH_, seed + 1);
    // Wv = W @ v_model
    gemm_wmma<0><<<dim3(TB_ / 128, NH_ / 128), 256, 0, stream>>>(
        W16, NV_, vm16, TB_, Wv, TB_, nullptr, 0, nullptr, 0, nullptr,
        NH_, TB_, NV_, 0u);
    // h_model = bernoulli(parallel_r(v_model, r_data))
    hsample_kernel<<<(int)(((size_t)NH_ * TB_) / 256), 256, 0, stream>>>(
        Wv, Ur, b_init, b_h, h16, seed + 2);
    seed += 16;
  }

  // Final v_model = bernoulli(sigmoid(W^T @ h + b_v)) -> f32 straight to d_out
  gemm_wmma<1><<<dim3(TB_ / 128, NV_ / 128), 256, 0, stream>>>(
      WT16, NH_, h16, TB_, v_model_out, TB_, nullptr, 0, nullptr, 0, b_v,
      NV_, TB_, NH_, 0xBEEFu);
}
